// Decoder_7464653160634
// MI455X (gfx1250) — matrix operand order repaired and checked by reading, not compiled, not run on hardware
//
#include <hip/hip_runtime.h>
#include <math.h>

// ---------------------------------------------------------------------------
// Tacotron2 decoder for MI455X (gfx1250).
// Strategy: bf16 WMMA (v_wmma_f32_16x16x32_bf16) for both LSTM GEMMs,
// weights converted once to bf16 (36MB, L2-resident on 192MB L2),
// one persistent kernel with a global barrier walks the 64 sequential steps.
// ---------------------------------------------------------------------------

typedef __bf16 v8bf  __attribute__((ext_vector_type(8)));
typedef __bf16 v16bf __attribute__((ext_vector_type(16)));
typedef float  v8f   __attribute__((ext_vector_type(8)));

#define NBLK 32
#define NTHR 256
#define GSTR (NBLK * NTHR)

#define BB    16
#define TENC  256
#define TDEC  64
#define NMEL  80
#define EDIM  512
#define RNN   1024
#define PREN  256
#define ADIM  128
#define LFILT 32
#define LKER  31

// ---- workspace layout (bytes) ---------------------------------------------
static constexpr size_t SZ_WATT_IH = (size_t)4096 * 768 * 2;
static constexpr size_t SZ_WATT_HH = (size_t)4096 * 1024 * 2;
static constexpr size_t SZ_WDEC_IH = (size_t)4096 * 1536 * 2;
static constexpr size_t SZ_WDEC_HH = (size_t)4096 * 1024 * 2;
static constexpr size_t SZ_PRE_BF  = (size_t)TDEC * BB * PREN * 2;
static constexpr size_t SZ_PM      = (size_t)BB * TENC * ADIM * 4;
static constexpr size_t SZ_H1      = (size_t)TDEC * BB * PREN * 4;
static constexpr size_t SZ_GATES   = (size_t)BB * 4096 * 4;

static constexpr size_t OFF_WATT_IH = 0;
static constexpr size_t OFF_WATT_HH = OFF_WATT_IH + SZ_WATT_IH;
static constexpr size_t OFF_WDEC_IH = OFF_WATT_HH + SZ_WATT_HH;
static constexpr size_t OFF_WDEC_HH = OFF_WDEC_IH + SZ_WDEC_IH;
static constexpr size_t OFF_PRE_BF  = OFF_WDEC_HH + SZ_WDEC_HH;
static constexpr size_t OFF_PM      = OFF_PRE_BF + SZ_PRE_BF;
static constexpr size_t OFF_H1      = OFF_PM + SZ_PM;
static constexpr size_t OFF_GATES   = OFF_H1 + SZ_H1;
// zero-initialized state region starts here
static constexpr size_t OFF_AH      = OFF_GATES + SZ_GATES;
static constexpr size_t OFF_AC      = OFF_AH + (size_t)BB * RNN * 4;
static constexpr size_t OFF_DH      = OFF_AC + (size_t)BB * RNN * 4;
static constexpr size_t OFF_DC      = OFF_DH + (size_t)BB * RNN * 4;
static constexpr size_t OFF_CTX     = OFF_DC + (size_t)BB * RNN * 4;
static constexpr size_t OFF_AW      = OFF_CTX + (size_t)BB * EDIM * 4;
static constexpr size_t OFF_AWCUM   = OFF_AW + (size_t)BB * TENC * 4;
static constexpr size_t OFF_QUERY   = OFF_AWCUM + (size_t)BB * TENC * 4;
static constexpr size_t OFF_ENERGY  = OFF_QUERY + (size_t)BB * ADIM * 4;
static constexpr size_t OFF_AHBF    = OFF_ENERGY + (size_t)BB * TENC * 4;
static constexpr size_t OFF_DHBF    = OFF_AHBF + (size_t)BB * RNN * 2;
static constexpr size_t OFF_CTXBF   = OFF_DHBF + (size_t)BB * RNN * 2;
static constexpr size_t OFF_SYNC    = OFF_CTXBF + (size_t)BB * EDIM * 2;
static constexpr size_t OFF_END     = OFF_SYNC + 256;
static constexpr size_t ZERO_BYTES  = OFF_END - OFF_AH;   // state + sync

// ---- WMMA helpers ----------------------------------------------------------
// A fragment: 16x32 bf16, rows (M) in lanes 0-15 / 16-31, split-K per ISA
// layout: lane<16 holds K[0..7],K[16..23]; lane>=16 holds K[8..15],K[24..31].
__device__ __forceinline__ v16bf frag_a(const __bf16* base, int ld, int k0, int lane) {
  int r  = lane & 15;
  int ks = (lane >> 4) << 3;            // 0 or 8
  const __bf16* p = base + (size_t)r * ld + k0 + ks;
  v8bf lo = *(const v8bf*)p;            // K k0+ks .. +7
  v8bf hi = *(const v8bf*)(p + 16);     // K k0+16+ks .. +7
  return __builtin_shufflevector(lo, hi, 0,1,2,3,4,5,6,7,8,9,10,11,12,13,14,15);
}

// B fragment: 32x16 bf16 (KxN) where column n = weight row (n0+n), split-K per ISA
// layout: lane<16 holds K[0..7],K[16..23]; lane>=16 holds K[8..15],K[24..31].
__device__ __forceinline__ v16bf frag_b(const __bf16* wbase, int ld, int n0, int k0, int lane) {
  int n  = lane & 15;
  int kh = (lane >> 4) << 3;            // 0 or 8
  const __bf16* p = wbase + (size_t)(n0 + n) * ld + k0 + kh;
  v8bf lo = *(const v8bf*)p;
  v8bf hi = *(const v8bf*)(p + 16);
  return __builtin_shufflevector(lo, hi, 0,1,2,3,4,5,6,7,8,9,10,11,12,13,14,15);
}

__device__ __forceinline__ v8f wmma_bf16(v16bf a, v16bf b, v8f c) {
  return __builtin_amdgcn_wmma_f32_16x16x32_bf16(false, a, false, b, (short)0, c, false, false);
}

__device__ __forceinline__ float sigf(float x) { return 1.0f / (1.0f + __expf(-x)); }

// ---- global barrier (persistent kernel, NBLK blocks all resident) ----------
__device__ __forceinline__ void gsync(unsigned* cnt, unsigned* gen) {
  __syncthreads();
  if (threadIdx.x == 0) {
    __threadfence();
    unsigned g = __hip_atomic_load(gen, __ATOMIC_RELAXED, __HIP_MEMORY_SCOPE_AGENT);
    unsigned v = __hip_atomic_fetch_add(cnt, 1u, __ATOMIC_RELAXED, __HIP_MEMORY_SCOPE_AGENT);
    if (v == NBLK - 1) {
      __hip_atomic_store(cnt, 0u, __ATOMIC_RELAXED, __HIP_MEMORY_SCOPE_AGENT);
      __threadfence();
      __hip_atomic_fetch_add(gen, 1u, __ATOMIC_RELAXED, __HIP_MEMORY_SCOPE_AGENT);
    } else {
      while (__hip_atomic_load(gen, __ATOMIC_RELAXED, __HIP_MEMORY_SCOPE_AGENT) == g)
        __builtin_amdgcn_s_sleep(1);
    }
    __threadfence();
  }
  __syncthreads();
}

// ---- prep kernels ----------------------------------------------------------
__global__ void k_convert(const float* __restrict__ src, __bf16* __restrict__ dst, int n) {
  for (int i = blockIdx.x * blockDim.x + threadIdx.x; i < n; i += gridDim.x * blockDim.x)
    dst[i] = (__bf16)src[i];
}

__global__ void k_zero(unsigned* __restrict__ dst, int nwords) {
  for (int i = blockIdx.x * blockDim.x + threadIdx.x; i < nwords; i += gridDim.x * blockDim.x)
    dst[i] = 0u;
}

// prenet layer 1: h1[t,b,j] = relu(b1[j] + W1[j,:] . frame(t,b,:))
// frame(0)=go(zeros), frame(t)=decoder_inputs[b,:,t-1]
__global__ void k_prenet1(const float* __restrict__ din, const float* __restrict__ W1,
                          const float* __restrict__ b1, float* __restrict__ h1) {
  const int N = TDEC * BB * PREN;
  for (int idx = blockIdx.x * blockDim.x + threadIdx.x; idx < N; idx += gridDim.x * blockDim.x) {
    int j = idx % PREN;
    int b = (idx / PREN) % BB;
    int t = idx / (PREN * BB);
    float s = b1[j];
    if (t > 0) {
      const float* x = din + (size_t)b * NMEL * TDEC + (t - 1);  // stride TDEC over mel
      const float* w = W1 + (size_t)j * NMEL;
      for (int m = 0; m < NMEL; ++m) s += w[m] * x[(size_t)m * TDEC];
    }
    h1[idx] = fmaxf(s, 0.0f);
  }
}

// prenet layer 2 -> bf16
__global__ void k_prenet2(const float* __restrict__ h1, const float* __restrict__ W2,
                          const float* __restrict__ b2, __bf16* __restrict__ prebf) {
  const int N = TDEC * BB * PREN;
  for (int idx = blockIdx.x * blockDim.x + threadIdx.x; idx < N; idx += gridDim.x * blockDim.x) {
    int j  = idx % PREN;
    int tb = idx / PREN;
    const float* x = h1 + (size_t)tb * PREN;
    const float* w = W2 + (size_t)j * PREN;
    float s = b2[j];
    for (int k = 0; k < PREN; ++k) s += w[k] * x[k];
    prebf[idx] = (__bf16)fmaxf(s, 0.0f);
  }
}

// processed_memory[b,te,d] = memory[b,te,:] . mem_W[d,:]
__global__ void k_pm(const float* __restrict__ mem, const float* __restrict__ mW,
                     float* __restrict__ pm) {
  const int N = BB * TENC * ADIM;
  for (int idx = blockIdx.x * blockDim.x + threadIdx.x; idx < N; idx += gridDim.x * blockDim.x) {
    int d  = idx & (ADIM - 1);
    int bt = idx >> 7;
    const float* x = mem + (size_t)bt * EDIM;
    const float* w = mW + (size_t)d * EDIM;
    float s = 0.0f;
    for (int k = 0; k < EDIM; ++k) s += w[k] * x[k];
    pm[idx] = s;
  }
}

// ---- persistent decode kernel ----------------------------------------------
struct DecParams {
  const float* memory;      // (16,256,512) f32
  const int*   mlen;        // (16,)
  const float* abih; const float* abhh;
  const float* qW;          // (128,1024)
  const float* convW;       // (32,2,31)
  const float* ldW;         // (128,32)
  const float* av;          // (1,128)
  const float* dbih; const float* dbhh;
  const float* projW;       // (80,1536)
  const float* projb;       // (80,)
  const float* gateW;       // (1,1536)
  const float* gateb;       // (1,)
  float* out_mel;           // (16,80,64)
  float* out_gate;          // (16,64)
  float* out_align;         // (16,64,256)
  char*  ws;
};

__global__ void __launch_bounds__(NTHR, 1) k_decode(DecParams P) {
  char* ws = P.ws;
  __bf16* w_att_ih = (__bf16*)(ws + OFF_WATT_IH);
  __bf16* w_att_hh = (__bf16*)(ws + OFF_WATT_HH);
  __bf16* w_dec_ih = (__bf16*)(ws + OFF_WDEC_IH);
  __bf16* w_dec_hh = (__bf16*)(ws + OFF_WDEC_HH);
  __bf16* pre_bf   = (__bf16*)(ws + OFF_PRE_BF);
  float*  pm       = (float*)(ws + OFF_PM);
  float*  gates    = (float*)(ws + OFF_GATES);
  float*  ah = (float*)(ws + OFF_AH);
  float*  ac = (float*)(ws + OFF_AC);
  float*  dh = (float*)(ws + OFF_DH);
  float*  dc = (float*)(ws + OFF_DC);
  float*  ctx = (float*)(ws + OFF_CTX);
  float*  aw = (float*)(ws + OFF_AW);
  float*  awcum = (float*)(ws + OFF_AWCUM);
  float*  query = (float*)(ws + OFF_QUERY);
  float*  energy = (float*)(ws + OFF_ENERGY);
  __bf16* ah_bf  = (__bf16*)(ws + OFF_AHBF);
  __bf16* dh_bf  = (__bf16*)(ws + OFF_DHBF);
  __bf16* ctx_bf = (__bf16*)(ws + OFF_CTXBF);
  unsigned* cnt = (unsigned*)(ws + OFF_SYNC);
  unsigned* gen = cnt + 16;

  const int gtid  = blockIdx.x * NTHR + threadIdx.x;
  const int lane  = threadIdx.x & 31;
  const int gwave = gtid >> 5;   // 0..255: one 16-wide N-tile per wave

  for (int t = 0; t < TDEC; ++t) {
    // ---- Phase A: attn-LSTM gates(16x4096) = [pre_t|ctx] @ WihT + ah @ WhhT
    for (int nt = gwave; nt < 256; nt += GSTR / 32) {
      const int n0 = nt * 16;
      v8f acc = {};
      const __bf16* preT = pre_bf + (size_t)t * BB * PREN;
      #pragma unroll 4
      for (int k0 = 0; k0 < PREN; k0 += 32) {
        acc = wmma_bf16(frag_a(preT, PREN, k0, lane),
                        frag_b(w_att_ih, 768, n0, k0, lane), acc);
      }
      #pragma unroll 4
      for (int k0 = 0; k0 < EDIM; k0 += 32) {
        acc = wmma_bf16(frag_a(ctx_bf, EDIM, k0, lane),
                        frag_b(w_att_ih, 768, n0, PREN + k0, lane), acc);
      }
      #pragma unroll 4
      for (int k0 = 0; k0 < RNN; k0 += 32) {
        acc = wmma_bf16(frag_a(ah_bf, RNN, k0, lane),
                        frag_b(w_att_hh, RNN, n0, k0, lane), acc);
      }
      const int n  = n0 + (lane & 15);
      const int mb = (lane >> 4) * 8;
      #pragma unroll
      for (int g = 0; g < 8; ++g) gates[(size_t)(mb + g) * 4096 + n] = acc[g];
    }
    gsync(cnt, gen);

    // ---- Phase B: attn-LSTM activations -> ah, ac
    for (int idx = gtid; idx < BB * RNN; idx += GSTR) {
      int b = idx >> 10, n = idx & (RNN - 1);
      const float* gr = gates + (size_t)b * 4096;
      float gi = gr[n]        + P.abih[n]        + P.abhh[n];
      float gf = gr[n + 1024] + P.abih[n + 1024] + P.abhh[n + 1024];
      float gg = gr[n + 2048] + P.abih[n + 2048] + P.abhh[n + 2048];
      float go = gr[n + 3072] + P.abih[n + 3072] + P.abhh[n + 3072];
      float c2 = sigf(gf) * ac[idx] + sigf(gi) * tanhf(gg);
      float h2 = sigf(go) * tanhf(c2);
      ac[idx] = c2; ah[idx] = h2; ah_bf[idx] = (__bf16)h2;
    }
    gsync(cnt, gen);

    // ---- Phase C1: query = ah @ query_W.T  (16x128)
    for (int idx = gtid; idx < BB * ADIM; idx += GSTR) {
      int b = idx >> 7, d = idx & (ADIM - 1);
      const float* h = ah + (size_t)b * RNN;
      const float* w = P.qW + (size_t)d * RNN;
      float s = 0.0f;
      for (int k = 0; k < RNN; ++k) s += w[k] * h[k];
      query[idx] = s;
    }
    gsync(cnt, gen);

    // ---- Phase C2: location conv + dense + energies (16x256)
    for (int idx = gtid; idx < BB * TENC; idx += GSTR) {
      int b = idx >> 8, te = idx & (TENC - 1);
      float locf[LFILT];
      #pragma unroll
      for (int f = 0; f < LFILT; ++f) locf[f] = 0.0f;
      for (int k = 0; k < LKER; ++k) {
        int p = te + k - (LKER - 1) / 2;
        if (p < 0 || p >= TENC) continue;
        float a0 = aw[b * TENC + p];
        float a1 = awcum[b * TENC + p];
        #pragma unroll
        for (int f = 0; f < LFILT; ++f)
          locf[f] += P.convW[f * 62 + k] * a0 + P.convW[f * 62 + 31 + k] * a1;
      }
      float e = 0.0f;
      for (int d = 0; d < ADIM; ++d) {
        float s = query[b * ADIM + d] + pm[(size_t)idx * ADIM + d];
        #pragma unroll 8
        for (int f = 0; f < LFILT; ++f) s += P.ldW[d * LFILT + f] * locf[f];
        e += P.av[d] * tanhf(s);
      }
      if (te >= P.mlen[b]) e = -1.0e9f;
      energy[idx] = e;
    }
    gsync(cnt, gen);

    // ---- Phase C3: masked softmax over 256, one wave per batch row
    if (gwave < BB) {
      int b = gwave;
      float vals[8], vmax = -3.0e38f;
      #pragma unroll
      for (int i = 0; i < 8; ++i) {
        vals[i] = energy[b * TENC + lane + i * 32];
        vmax = fmaxf(vmax, vals[i]);
      }
      for (int m = 16; m >= 1; m >>= 1) vmax = fmaxf(vmax, __shfl_xor(vmax, m));
      float ssum = 0.0f;
      #pragma unroll
      for (int i = 0; i < 8; ++i) { vals[i] = __expf(vals[i] - vmax); ssum += vals[i]; }
      for (int m = 16; m >= 1; m >>= 1) ssum += __shfl_xor(ssum, m);
      float inv = 1.0f / ssum;
      #pragma unroll
      for (int i = 0; i < 8; ++i) {
        int te = lane + i * 32;
        float w = vals[i] * inv;
        aw[b * TENC + te] = w;
        awcum[b * TENC + te] += w;
        P.out_align[(size_t)b * TDEC * TENC + (size_t)t * TENC + te] = w;
      }
    }
    gsync(cnt, gen);

    // ---- Phase C4: ctx[b,d] = sum_te aw * memory
    for (int idx = gtid; idx < BB * EDIM; idx += GSTR) {
      int b = idx >> 9, d = idx & (EDIM - 1);
      const float* m = P.memory + (size_t)b * TENC * EDIM + d;
      const float* w = aw + b * TENC;
      float s = 0.0f;
      for (int te = 0; te < TENC; ++te) s += w[te] * m[(size_t)te * EDIM];
      ctx[idx] = s; ctx_bf[idx] = (__bf16)s;
    }
    gsync(cnt, gen);

    // ---- Phase D: dec-LSTM gates(16x4096) = [ah|ctx] @ WihT + dh @ WhhT
    for (int nt = gwave; nt < 256; nt += GSTR / 32) {
      const int n0 = nt * 16;
      v8f acc = {};
      #pragma unroll 4
      for (int k0 = 0; k0 < RNN; k0 += 32) {
        acc = wmma_bf16(frag_a(ah_bf, RNN, k0, lane),
                        frag_b(w_dec_ih, 1536, n0, k0, lane), acc);
      }
      #pragma unroll 4
      for (int k0 = 0; k0 < EDIM; k0 += 32) {
        acc = wmma_bf16(frag_a(ctx_bf, EDIM, k0, lane),
                        frag_b(w_dec_ih, 1536, n0, RNN + k0, lane), acc);
      }
      #pragma unroll 4
      for (int k0 = 0; k0 < RNN; k0 += 32) {
        acc = wmma_bf16(frag_a(dh_bf, RNN, k0, lane),
                        frag_b(w_dec_hh, RNN, n0, k0, lane), acc);
      }
      const int n  = n0 + (lane & 15);
      const int mb = (lane >> 4) * 8;
      #pragma unroll
      for (int g = 0; g < 8; ++g) gates[(size_t)(mb + g) * 4096 + n] = acc[g];
    }
    gsync(cnt, gen);

    // ---- Phase E: dec-LSTM activations -> dh, dc
    for (int idx = gtid; idx < BB * RNN; idx += GSTR) {
      int b = idx >> 10, n = idx & (RNN - 1);
      const float* gr = gates + (size_t)b * 4096;
      float gi = gr[n]        + P.dbih[n]        + P.dbhh[n];
      float gf = gr[n + 1024] + P.dbih[n + 1024] + P.dbhh[n + 1024];
      float gg = gr[n + 2048] + P.dbih[n + 2048] + P.dbhh[n + 2048];
      float go = gr[n + 3072] + P.dbih[n + 3072] + P.dbhh[n + 3072];
      float c2 = sigf(gf) * dc[idx] + sigf(gi) * tanhf(gg);
      float h2 = sigf(go) * tanhf(c2);
      dc[idx] = c2; dh[idx] = h2; dh_bf[idx] = (__bf16)h2;
    }
    gsync(cnt, gen);

    // ---- Phase F: projection + gate from [dh|ctx]
    for (int idx = gtid; idx < BB * (NMEL + 1); idx += GSTR) {
      int b = idx / (NMEL + 1), j = idx % (NMEL + 1);
      const float* w = (j < NMEL) ? (P.projW + (size_t)j * 1536) : P.gateW;
      const float* h = dh + (size_t)b * RNN;
      const float* c = ctx + (size_t)b * EDIM;
      float s = 0.0f;
      for (int k = 0; k < RNN; ++k) s += w[k] * h[k];
      for (int k = 0; k < EDIM; ++k) s += w[RNN + k] * c[k];
      if (j < NMEL) P.out_mel[(size_t)b * NMEL * TDEC + (size_t)j * TDEC + t] = s + P.projb[j];
      else          P.out_gate[b * TDEC + t] = s + P.gateb[0];
    }
    gsync(cnt, gen);
  }
}

// ---------------------------------------------------------------------------
extern "C" void kernel_launch(void* const* d_in, const int* in_sizes, int n_in,
                              void* d_out, int out_size, void* d_ws, size_t ws_size,
                              hipStream_t stream) {
  const float* memory = (const float*)d_in[0];
  const float* dinp   = (const float*)d_in[1];
  const int*   mlen   = (const int*)d_in[2];
  const float* pW1 = (const float*)d_in[3];
  const float* pb1 = (const float*)d_in[4];
  const float* pW2 = (const float*)d_in[5];
  const float* pb2 = (const float*)d_in[6];
  const float* aWih = (const float*)d_in[7];
  const float* aWhh = (const float*)d_in[8];
  const float* abih = (const float*)d_in[9];
  const float* abhh = (const float*)d_in[10];
  const float* qW   = (const float*)d_in[11];
  const float* mW   = (const float*)d_in[12];
  const float* convW = (const float*)d_in[13];
  const float* ldW   = (const float*)d_in[14];
  const float* av    = (const float*)d_in[15];
  const float* dWih = (const float*)d_in[16];
  const float* dWhh = (const float*)d_in[17];
  const float* dbih = (const float*)d_in[18];
  const float* dbhh = (const float*)d_in[19];
  const float* projW = (const float*)d_in[20];
  const float* projb = (const float*)d_in[21];
  const float* gateW = (const float*)d_in[22];
  const float* gateb = (const float*)d_in[23];

  char*  ws  = (char*)d_ws;
  float* out = (float*)d_out;
  float* out_mel   = out;                     // 16*80*64
  float* out_gate  = out + 16 * 80 * 64;      // 16*64
  float* out_align = out_gate + 16 * 64;      // 16*64*256

  // weight fp32 -> bf16 (once per launch; L2-resident afterwards)
  k_convert<<<dim3(512), dim3(256), 0, stream>>>(aWih, (__bf16*)(ws + OFF_WATT_IH), 4096 * 768);
  k_convert<<<dim3(512), dim3(256), 0, stream>>>(aWhh, (__bf16*)(ws + OFF_WATT_HH), 4096 * 1024);
  k_convert<<<dim3(512), dim3(256), 0, stream>>>(dWih, (__bf16*)(ws + OFF_WDEC_IH), 4096 * 1536);
  k_convert<<<dim3(512), dim3(256), 0, stream>>>(dWhh, (__bf16*)(ws + OFF_WDEC_HH), 4096 * 1024);
  // zero LSTM/attention state + barrier counters
  k_zero<<<dim3(128), dim3(256), 0, stream>>>((unsigned*)(ws + OFF_AH), (int)(ZERO_BYTES / 4));
  // prenet (two dense layers) + processed memory
  k_prenet1<<<dim3(512), dim3(256), 0, stream>>>(dinp, pW1, pb1, (float*)(ws + OFF_H1));
  k_prenet2<<<dim3(512), dim3(256), 0, stream>>>((float*)(ws + OFF_H1), pW2, pb2,
                                                 (__bf16*)(ws + OFF_PRE_BF));
  k_pm<<<dim3(1024), dim3(256), 0, stream>>>(memory, mW, (float*)(ws + OFF_PM));

  DecParams P;
  P.memory = memory; P.mlen = mlen;
  P.abih = abih; P.abhh = abhh;
  P.qW = qW; P.convW = convW; P.ldW = ldW; P.av = av;
  P.dbih = dbih; P.dbhh = dbhh;
  P.projW = projW; P.projb = projb; P.gateW = gateW; P.gateb = gateb;
  P.out_mel = out_mel; P.out_gate = out_gate; P.out_align = out_align;
  P.ws = ws;

  k_decode<<<dim3(NBLK), dim3(NTHR), 0, stream>>>(P);
}
